// VNN_DGCNN_37795712205046
// MI455X (gfx1250) — matrix operand order repaired and checked by reading, not compiled, not run on hardware
//
#include <hip/hip_runtime.h>
#include <hip/hip_bf16.h>

typedef __attribute__((ext_vector_type(16))) _Float16 v16h;
typedef __attribute__((ext_vector_type(8)))  _Float16 v8h;
typedef __attribute__((ext_vector_type(8)))  float    v8f;

#define NB    8
#define NPTS  1024
#define KNN   20
#define HCH   64
#define COLS  (NB * 3 * NPTS)   // 24576

// ---------------- WMMA fragment loaders (ISA 7.12.2 layouts) ----------------
// A 16x32 f16: lane L holds row m=L&15. lanes<16: K in {0..7} then {16..23};
// lanes>=16: K in {8..15} then {24..31}. Row data is K-contiguous in memory.
__device__ __forceinline__ v16h load_a_frag(const _Float16* rowbase, int lane) {
  int hi = (lane < 16) ? 0 : 8;
  v8h lo = *(const v8h*)(rowbase + hi);
  v8h up = *(const v8h*)(rowbase + 16 + hi);
  v16h r;
#pragma unroll
  for (int e = 0; e < 8; ++e) { r[e] = lo[e]; r[e + 8] = up[e]; }
  return r;
}
// B 32x16 f16: lane L holds column n=L&15; lanes<16: K in {0..7} then {16..23}; lanes>=16: K in {8..15} then {24..31}.
__device__ __forceinline__ v16h load_b_frag(const _Float16* colbase, int lane) {
  int ko = (lane < 16) ? 0 : 8;
  v8h lo = *(const v8h*)(colbase + ko);
  v8h up = *(const v8h*)(colbase + 16 + ko);
  v16h r;
#pragma unroll
  for (int e = 0; e < 8; ++e) { r[e] = lo[e]; r[e + 8] = up[e]; }
  return r;
}

__device__ __forceinline__ float wave_sum32(float v) {
#pragma unroll
  for (int off = 16; off >= 1; off >>= 1) v += __shfl_xor(v, off, 32);
  return v;
}

// ---------------- conversion kernels ----------------
// Layer 1: x [B,N,3] -> flat16 [B,N,32] (pad), bmat [B,3,N,32], xx [B,N]
__global__ void cvt1_kernel(const float* __restrict__ x, _Float16* __restrict__ flat16,
                            _Float16* __restrict__ bmat, float* __restrict__ xx) {
  int i = blockIdx.x * blockDim.x + threadIdx.x;
  if (i >= NB * NPTS) return;
  int b = i >> 10, n = i & 1023;
  float v0 = x[i * 3 + 0], v1 = x[i * 3 + 1], v2 = x[i * 3 + 2];
  xx[i] = v0 * v0 + v1 * v1 + v2 * v2;
  _Float16* f = flat16 + (size_t)i * 32;
#pragma unroll
  for (int d = 0; d < 32; ++d) f[d] = (_Float16)0.0f;
  f[0] = (_Float16)v0; f[1] = (_Float16)v1; f[2] = (_Float16)v2;
  float vv[3] = {v0, v1, v2};
  for (int c3 = 0; c3 < 3; ++c3) {
    _Float16* bp = bmat + ((size_t)(b * 3 + c3) * NPTS + n) * 32;
#pragma unroll
    for (int c = 0; c < 32; ++c) bp[c] = (_Float16)0.0f;
    bp[0] = (_Float16)vv[c3];
  }
}

// Layers 2-4: xin [B,64,3,N] -> flat16 [B,N,192], bmat [B,3,N,64], xx [B,N]
__global__ void cvtX_kernel(const float* __restrict__ xin, _Float16* __restrict__ flat16,
                            _Float16* __restrict__ bmat, float* __restrict__ xx) {
  int i = blockIdx.x * blockDim.x + threadIdx.x;
  if (i >= NB * NPTS) return;
  int b = i >> 10, n = i & 1023;
  float ss = 0.0f;
  for (int h = 0; h < HCH; ++h) {
#pragma unroll
    for (int c3 = 0; c3 < 3; ++c3) {
      float v = xin[(((size_t)b * HCH + h) * 3 + c3) * NPTS + n];
      ss += v * v;
      flat16[((size_t)b * NPTS + n) * 192 + h * 3 + c3] = (_Float16)v;
      bmat[((size_t)(b * 3 + c3) * NPTS + n) * 64 + h] = (_Float16)v;
    }
  }
  xx[i] = ss;
}

// Final: cat(x1..x4) -> bmat [B,3,N,256]
__global__ void cvt_cat_kernel(const float* __restrict__ x1, const float* __restrict__ x2,
                               const float* __restrict__ x3, const float* __restrict__ x4,
                               _Float16* __restrict__ bmat) {
  int i = blockIdx.x * blockDim.x + threadIdx.x;
  if (i >= NB * NPTS) return;
  int b = i >> 10, n = i & 1023;
  const float* xs[4] = {x1, x2, x3, x4};
  for (int l = 0; l < 4; ++l)
    for (int h = 0; h < HCH; ++h)
#pragma unroll
      for (int c3 = 0; c3 < 3; ++c3)
        bmat[((size_t)(b * 3 + c3) * NPTS + n) * 256 + l * HCH + h] =
            (_Float16)xs[l][(((size_t)b * HCH + h) * 3 + c3) * NPTS + n];
}

// W4 [256, Cpad] f16: rows 0-63 wf_A, 64-127 wf_B - wf_A, 128-191 wd_A, 192-255 wd_B - wd_A
__global__ void build_w4_kernel(const float* __restrict__ wf, const float* __restrict__ wd,
                                _Float16* __restrict__ W4, int C, int Cpad) {
  int r = threadIdx.x;
  int g = r >> 6, o = r & 63;
  const float* w = (g < 2) ? wf : wd;
  int gg = g & 1;
  _Float16* dst = W4 + (size_t)r * Cpad;
  for (int c = 0; c < Cpad; ++c) {
    float v = 0.0f;
    if (c < C) v = gg ? (w[o * 2 * C + C + c] - w[o * 2 * C + c]) : w[o * 2 * C + c];
    dst[c] = (_Float16)v;
  }
}

__global__ void build_wc_kernel(const float* __restrict__ wcf, _Float16* __restrict__ Wc) {
  int r = threadIdx.x;  // 128
  for (int c = 0; c < 256; ++c) Wc[(size_t)r * 256 + c] = (_Float16)wcf[r * 256 + c];
}

// ---------------- kNN: WMMA Gram + iterative top-k ----------------
__global__ __launch_bounds__(256) void gram_topk_kernel(const _Float16* __restrict__ flat16,
                                                        const float* __restrict__ xx,
                                                        int* __restrict__ idx_out, int Dpad) {
  __shared__ float sc[16][NPTS];  // 64 KB
  int blk = blockIdx.x;
  int b = blk >> 6, rt = blk & 63;
  int row0 = rt * 16;
  int tid = threadIdx.x, w = tid >> 5, lane = tid & 31;
  const _Float16* base = flat16 + (size_t)b * NPTS * Dpad;
  int mrow = row0 + (lane & 15);
  const _Float16* arow = base + (size_t)mrow * Dpad;

  for (int ct = w; ct < 64; ct += 8) {
    int col = ct * 16 + (lane & 15);
    const _Float16* bcol = base + (size_t)col * Dpad;
    v8f acc = {};
    for (int kb = 0; kb < Dpad; kb += 32) {
      v16h a = load_a_frag(arow + kb, lane);
      v16h bf = load_b_frag(bcol + kb, lane);
      acc = __builtin_amdgcn_wmma_f32_16x16x32_f16(false, a, false, bf, (short)0, acc,
                                                   false, false);
    }
    float xxc = xx[b * NPTS + col];
    int rbase = (lane < 16) ? 0 : 8;
#pragma unroll
    for (int v = 0; v < 8; ++v) {
      int r = rbase + v;
      sc[r][col] = 2.0f * acc[v] - xx[b * NPTS + row0 + r] - xxc;
    }
  }
  __syncthreads();

  // top-20 per row: group g (16 lanes) owns row g; stride-16 scan is bank-conflict free
  int g = tid >> 4, sub = tid & 15;
  for (int t = 0; t < KNN; ++t) {
    float bestv = -3.0e38f;
    int besti = sub;
    for (int c = sub; c < NPTS; c += 16) {
      float vv = sc[g][c];
      if (vv > bestv || (vv == bestv && c < besti)) { bestv = vv; besti = c; }
    }
#pragma unroll
    for (int off = 1; off < 16; off <<= 1) {
      float ov = __shfl_xor(bestv, off, 32);
      int oi = __shfl_xor(besti, off, 32);
      if (ov > bestv || (ov == bestv && oi < besti)) { bestv = ov; besti = oi; }
    }
    if (sub == 0) {
      idx_out[((size_t)b * NPTS + row0 + g) * KNN + t] = besti;
      sc[g][besti] = -3.0e38f;
    }
    __syncthreads();
  }
}

// ---------------- generic WMMA GEMM: C[M,cols] = A[M,Kp] * B^T[cols,Kp] ----------------
__global__ __launch_bounds__(256) void gemm16_kernel(const _Float16* __restrict__ A,
                                                     const _Float16* __restrict__ Bm,
                                                     float* __restrict__ C, int Kp, int cols) {
  int tid = threadIdx.x, w = tid >> 5, lane = tid & 31;
  int m0 = blockIdx.y * 16;
  int col = blockIdx.x * 128 + w * 16 + (lane & 15);
  const _Float16* arow = A + (size_t)(m0 + (lane & 15)) * Kp;
  const _Float16* bcol = Bm + (size_t)col * Kp;
  v8f acc = {};
  for (int kb = 0; kb < Kp; kb += 32) {
    v16h a = load_a_frag(arow + kb, lane);
    v16h bf = load_b_frag(bcol + kb, lane);
    acc = __builtin_amdgcn_wmma_f32_16x16x32_f16(false, a, false, bf, (short)0, acc,
                                                 false, false);
  }
  int rbase = (lane < 16) ? 0 : 8;
#pragma unroll
  for (int v = 0; v < 8; ++v) C[(size_t)(m0 + rbase + v) * cols + col] = acc[v];
}

// Yt element accessor: Yt[r, b, c3, n] with cols = b*3072 + c3*1024 + n
#define YT(r, c3, j) Yt[((size_t)(r) * COLS) + (b * 3072) + ((c3) << 10) + (j)]

// ---------------- edge BN stats (sum, sumsq of ||p|| per channel) ----------------
__global__ __launch_bounds__(256) void edge_stats_kernel(const float* __restrict__ Yt,
                                                         const int* __restrict__ idx,
                                                         float* __restrict__ stats) {
  __shared__ int sidx[KNN];
  int bn = blockIdx.x;
  int b = bn >> 10, n = bn & 1023;
  int tid = threadIdx.x;
  if (tid < KNN) sidx[tid] = idx[(size_t)bn * KNN + tid];
  __syncthreads();
  int h = tid >> 2, q = tid & 3;
  float pc0 = YT(64 + h, 0, n), pc1 = YT(64 + h, 1, n), pc2 = YT(64 + h, 2, n);
  float accs = 0.0f, accq = 0.0f;
  for (int kk = q; kk < KNN; kk += 4) {
    int j = sidx[kk];
    float p0 = YT(h, 0, j) + pc0;
    float p1 = YT(h, 1, j) + pc1;
    float p2 = YT(h, 2, j) + pc2;
    float nrm = sqrtf(p0 * p0 + p1 * p1 + p2 * p2) + 1e-6f;
    accs += nrm;
    accq += nrm * nrm;
  }
  accs += __shfl_xor(accs, 1, 32); accs += __shfl_xor(accs, 2, 32);
  accq += __shfl_xor(accq, 1, 32); accq += __shfl_xor(accq, 2, 32);
  if (q == 0) { atomicAdd(&stats[h * 2], accs); atomicAdd(&stats[h * 2 + 1], accq); }
}

// ---------------- edge apply: VN-BN + directional LeakyReLU + mean over k ----------------
__global__ __launch_bounds__(64) void edge_apply_kernel(const float* __restrict__ Yt,
                                                        const int* __restrict__ idx,
                                                        const float* __restrict__ stats,
                                                        const float* __restrict__ bnw,
                                                        const float* __restrict__ bnb,
                                                        float* __restrict__ xout) {
  __shared__ int sidx[KNN];
  int bn = blockIdx.x;
  int b = bn >> 10, n = bn & 1023;
  int h = threadIdx.x;
  if (h < KNN) sidx[h] = idx[(size_t)bn * KNN + h];
  __syncthreads();
  const float cnt = (float)(NB * NPTS * KNN);
  float mean = stats[h * 2] / cnt;
  float var = stats[h * 2 + 1] / cnt - mean * mean;
  float rstd = rsqrtf(var + 1e-5f);
  float w = bnw[h], bb = bnb[h];
  float pc0 = YT(64 + h, 0, n), pc1 = YT(64 + h, 1, n), pc2 = YT(64 + h, 2, n);
  float dc0 = YT(192 + h, 0, n), dc1 = YT(192 + h, 1, n), dc2 = YT(192 + h, 2, n);
  float a0 = 0.0f, a1 = 0.0f, a2 = 0.0f;
  for (int kk = 0; kk < KNN; ++kk) {
    int j = sidx[kk];
    float p0 = YT(h, 0, j) + pc0, p1 = YT(h, 1, j) + pc1, p2 = YT(h, 2, j) + pc2;
    float d0 = YT(128 + h, 0, j) + dc0, d1 = YT(128 + h, 1, j) + dc1, d2 = YT(128 + h, 2, j) + dc2;
    float nrm = sqrtf(p0 * p0 + p1 * p1 + p2 * p2) + 1e-6f;
    float nbn = (nrm - mean) * rstd * w + bb;
    float scl = nbn / nrm;
    p0 *= scl; p1 *= scl; p2 *= scl;
    float dot = p0 * d0 + p1 * d1 + p2 * d2;
    float dsq = d0 * d0 + d1 * d1 + d2 * d2;
    float t = (dot >= 0.0f) ? 0.0f : dot / (dsq + 1e-6f);
    a0 += p0 - 0.8f * t * d0;
    a1 += p1 - 0.8f * t * d1;
    a2 += p2 - 0.8f * t * d2;
  }
  const float inv = 1.0f / (float)KNN;
  size_t o = (((size_t)b * HCH + h) * 3) * NPTS + n;
  xout[o] = a0 * inv;
  xout[o + NPTS] = a1 * inv;
  xout[o + 2 * NPTS] = a2 * inv;
}

// ---------------- final layer: d = wcd . cat, BN stats, apply + mean over N ----------------
__global__ void final_d_kernel(const float* __restrict__ wcd, const float* __restrict__ x1,
                               const float* __restrict__ x2, const float* __restrict__ x3,
                               const float* __restrict__ x4, float* __restrict__ Dfin) {
  int j = blockIdx.x * blockDim.x + threadIdx.x;
  if (j >= COLS) return;
  int b = j / 3072, rem = j % 3072;
  int c3 = rem >> 10, n = rem & 1023;
  const float* xs[4] = {x1, x2, x3, x4};
  float s = 0.0f;
  for (int l = 0; l < 4; ++l)
    for (int h = 0; h < HCH; ++h)
      s += wcd[l * HCH + h] * xs[l][(((size_t)b * HCH + h) * 3 + c3) * NPTS + n];
  Dfin[j] = s;
}

__global__ __launch_bounds__(256) void final_stats_kernel(const float* __restrict__ Pfin,
                                                          float* __restrict__ stats) {
  int h = blockIdx.x, tid = threadIdx.x;
  float s = 0.0f, q = 0.0f;
  for (int i = tid; i < NB * NPTS; i += 256) {
    int b = i >> 10, n = i & 1023;
    size_t o = ((size_t)h * COLS) + b * 3072 + n;
    float p0 = Pfin[o], p1 = Pfin[o + NPTS], p2 = Pfin[o + 2 * NPTS];
    float nrm = sqrtf(p0 * p0 + p1 * p1 + p2 * p2) + 1e-6f;
    s += nrm;
    q += nrm * nrm;
  }
  s = wave_sum32(s);
  q = wave_sum32(q);
  __shared__ float ws_[8], wq_[8];
  int wv = tid >> 5;
  if ((tid & 31) == 0) { ws_[wv] = s; wq_[wv] = q; }
  __syncthreads();
  if (tid == 0) {
    float S = 0.0f, Q = 0.0f;
    for (int i = 0; i < 8; ++i) { S += ws_[i]; Q += wq_[i]; }
    stats[h * 2] = S;
    stats[h * 2 + 1] = Q;
  }
}

__global__ __launch_bounds__(256) void final_apply_kernel(const float* __restrict__ Pfin,
                                                          const float* __restrict__ Dfin,
                                                          const float* __restrict__ stats,
                                                          const float* __restrict__ bnw,
                                                          const float* __restrict__ bnb,
                                                          float* __restrict__ out) {
  int bh = blockIdx.x;
  int b = bh >> 7, h = bh & 127;
  int tid = threadIdx.x;
  const float cnt = (float)(NB * NPTS);
  float mean = stats[h * 2] / cnt;
  float var = stats[h * 2 + 1] / cnt - mean * mean;
  float rstd = rsqrtf(var + 1e-5f);
  float w = bnw[h], bb = bnb[h];
  float a0 = 0.0f, a1 = 0.0f, a2 = 0.0f;
  for (int n = tid; n < NPTS; n += 256) {
    size_t o = ((size_t)h * COLS) + b * 3072 + n;
    float p0 = Pfin[o], p1 = Pfin[o + NPTS], p2 = Pfin[o + 2 * NPTS];
    size_t od = (size_t)b * 3072 + n;
    float d0 = Dfin[od], d1 = Dfin[od + NPTS], d2 = Dfin[od + 2 * NPTS];
    float nrm = sqrtf(p0 * p0 + p1 * p1 + p2 * p2) + 1e-6f;
    float nbn = (nrm - mean) * rstd * w + bb;
    float scl = nbn / nrm;
    p0 *= scl; p1 *= scl; p2 *= scl;
    float dot = p0 * d0 + p1 * d1 + p2 * d2;
    float dsq = d0 * d0 + d1 * d1 + d2 * d2;
    float t = (dot >= 0.0f) ? 0.0f : dot / (dsq + 1e-6f);
    a0 += p0 - 0.8f * t * d0;
    a1 += p1 - 0.8f * t * d1;
    a2 += p2 - 0.8f * t * d2;
  }
  a0 = wave_sum32(a0); a1 = wave_sum32(a1); a2 = wave_sum32(a2);
  __shared__ float r0[8], r1[8], r2[8];
  int wv = tid >> 5;
  if ((tid & 31) == 0) { r0[wv] = a0; r1[wv] = a1; r2[wv] = a2; }
  __syncthreads();
  if (tid == 0) {
    float s0 = 0.0f, s1 = 0.0f, s2 = 0.0f;
    for (int i = 0; i < 8; ++i) { s0 += r0[i]; s1 += r1[i]; s2 += r2[i]; }
    const float inv = 1.0f / (float)NPTS;
    size_t o = ((size_t)b * 128 + h) * 3;
    out[o] = s0 * inv;
    out[o + 1] = s1 * inv;
    out[o + 2] = s2 * inv;
  }
}

// ---------------- host launcher ----------------
extern "C" void kernel_launch(void* const* d_in, const int* in_sizes, int n_in,
                              void* d_out, int out_size, void* d_ws, size_t ws_size,
                              hipStream_t stream) {
  (void)in_sizes; (void)n_in; (void)out_size; (void)ws_size;
  const float* x = (const float*)d_in[0];
  const float* wf[4]  = {(const float*)d_in[1], (const float*)d_in[5],
                         (const float*)d_in[9], (const float*)d_in[13]};
  const float* wd[4]  = {(const float*)d_in[2], (const float*)d_in[6],
                         (const float*)d_in[10], (const float*)d_in[14]};
  const float* bnw[4] = {(const float*)d_in[3], (const float*)d_in[7],
                         (const float*)d_in[11], (const float*)d_in[15]};
  const float* bnb[4] = {(const float*)d_in[4], (const float*)d_in[8],
                         (const float*)d_in[12], (const float*)d_in[16]};
  const float* wcf = (const float*)d_in[17];
  const float* wcd = (const float*)d_in[18];
  const float* bncw = (const float*)d_in[19];
  const float* bncb = (const float*)d_in[20];

  char* ws = (char*)d_ws;
  size_t off = 0;
  auto alloc = [&](size_t bytes) -> void* {
    off = (off + 255) & ~(size_t)255;
    void* p = ws + off;
    off += bytes;
    return p;
  };
  _Float16* flat16 = (_Float16*)alloc((size_t)NB * NPTS * 192 * 2);
  _Float16* bmat   = (_Float16*)alloc((size_t)NB * 3 * NPTS * 256 * 2);
  float*    xx     = (float*)alloc((size_t)NB * NPTS * 4);
  int*      idx    = (int*)alloc((size_t)NB * NPTS * KNN * 4);
  _Float16* Wh     = (_Float16*)alloc((size_t)256 * 256 * 2);
  float*    Yt     = (float*)alloc((size_t)256 * COLS * 4);   // also reused as Pfin
  float*    xl[4];
  for (int l = 0; l < 4; ++l) xl[l] = (float*)alloc((size_t)NB * HCH * 3 * NPTS * 4);
  float*    Dfin   = (float*)alloc((size_t)COLS * 4);
  float*    stats  = (float*)alloc((size_t)256 * 2 * 4);

  const int ptBlocks = (NB * NPTS) / 256;  // 32

  for (int l = 0; l < 4; ++l) {
    int C = (l == 0) ? 1 : HCH;
    int Cpad = (l == 0) ? 32 : 64;
    int Dpad = (l == 0) ? 32 : 192;
    if (l == 0)
      cvt1_kernel<<<ptBlocks, 256, 0, stream>>>(x, flat16, bmat, xx);
    else
      cvtX_kernel<<<ptBlocks, 256, 0, stream>>>(xl[l - 1], flat16, bmat, xx);
    build_w4_kernel<<<1, 256, 0, stream>>>(wf[l], wd[l], Wh, C, Cpad);
    gram_topk_kernel<<<NB * 64, 256, 0, stream>>>(flat16, xx, idx, Dpad);
    gemm16_kernel<<<dim3(COLS / 128, 256 / 16), 256, 0, stream>>>(Wh, bmat, Yt, Cpad, COLS);
    hipMemsetAsync(stats, 0, HCH * 2 * sizeof(float), stream);
    edge_stats_kernel<<<NB * NPTS, 256, 0, stream>>>(Yt, idx, stats);
    edge_apply_kernel<<<NB * NPTS, 64, 0, stream>>>(Yt, idx, stats, bnw[l], bnb[l], xl[l]);
  }

  // final: p = wcf @ cat (WMMA), d = wcd @ cat, VN-BN + LeakyReLU + mean over N
  cvt_cat_kernel<<<ptBlocks, 256, 0, stream>>>(xl[0], xl[1], xl[2], xl[3], bmat);
  build_wc_kernel<<<1, 128, 0, stream>>>(wcf, Wh);
  gemm16_kernel<<<dim3(COLS / 128, 128 / 16), 256, 0, stream>>>(Wh, bmat, Yt, 256, COLS);
  final_d_kernel<<<COLS / 256, 256, 0, stream>>>(wcd, xl[0], xl[1], xl[2], xl[3], Dfin);
  final_stats_kernel<<<128, 256, 0, stream>>>(Yt, stats);
  final_apply_kernel<<<NB * 128, 256, 0, stream>>>(Yt, Dfin, stats, bncw, bncb,
                                                   (float*)d_out);
}
